// DenoisedNL_4320737099855
// MI455X (gfx1250) — compile-verified, matrix operand order checked by reading, not run on hardware
//
#include <hip/hip_runtime.h>
#include <hip/hip_bf16.h>

// ---------------------------------------------------------------------------
// CDNA5 (gfx1250) types for WMMA
// ---------------------------------------------------------------------------
typedef __attribute__((ext_vector_type(16))) __bf16 v16bf;
typedef __attribute__((ext_vector_type(8)))  float  v8f;

// Problem constants (match reference setup_inputs)
#define BB   2
#define CC   512
#define CQ   64
#define HH   48
#define WW   48
#define NN   (HH*WW)        // 2304
#define H2   96
#define W2   96
#define N2   (H2*W2)        // 9216
#define NCC  19
#define C2   1024

// GEMM tiling: 128x64 block tile, 4 waves, each wave 32(M)x64(N)
#define TM 128
#define TN 64
#define TK 32
#define GEMM_THREADS 128
#define ACH ((TM*TK)/8)     // 512 A chunks of 8 bf16
#define BCH ((TK*TN)/8)     // 256 B chunks of 8 bf16

// ---------------------------------------------------------------------------
// f32 -> bf16 convert
// ---------------------------------------------------------------------------
__global__ __launch_bounds__(256)
void k_cvt_bf16(const float* __restrict__ in, __bf16* __restrict__ out, int n) {
    int i = blockIdx.x * blockDim.x + threadIdx.x;
    if (i < n) out[i] = (__bf16)in[i];
}

// ---------------------------------------------------------------------------
// Fragment-layout helpers.
// ISA 7.12.2 (16-bit A 16x32): lane L = r + 16*((K>>3)&1) holds, contiguously,
//   pos 0..7  = row r, K in [khalf,   khalf+8)
//   pos 8..15 = row r, K in [16+khalf,16+khalf+8)
// so an 8-element K-run is one aligned 16B chunk of one lane's fragment.
// ---------------------------------------------------------------------------
__device__ __forceinline__ __bf16* frag_slot(v16bf* base, int s, int r, int kk0) {
    int lane = r + ((kk0 & 8) ? 16 : 0);
    int half = (kk0 & 16) ? 1 : 0;
    return (__bf16*)(base + (s * 32 + lane)) + half * 8;
}

// ---------------------------------------------------------------------------
// Generic stride-parameterized BF16 WMMA GEMM, fp32 accumulate.
//   C[b, m, n] (ldc row stride) = sum_k A[b*aBatch + m*am + k*ak]
//                                       * B[b*bBatch + k*bk + n*bn]
// All strides are 32-bit element counts (largest offset here ~10.6M < 2^31).
// Zero-pads out-of-range M/N/K (supports K=19 etc.).
// Double-buffered LDS staging directly in WMMA fragment layout.
// ---------------------------------------------------------------------------
__global__ __launch_bounds__(GEMM_THREADS)
void k_gemm_bf16(const __bf16* __restrict__ A, const __bf16* __restrict__ Bm,
                 float* __restrict__ C, __bf16* __restrict__ Cb,
                 int M, int Nn, int K,
                 int aBatch, int am, int ak,
                 int bBatch, int bk, int bn,
                 int cBatch, int ldc)
{
    __shared__ v16bf AsF[2][8 * 32];   // 8 M-subtiles x 32 lanes, x2 buffers
    __shared__ v16bf BsF[2][4 * 32];   // 4 N-subtiles x 32 lanes, x2 buffers

    const int b  = blockIdx.z;
    const int m0 = blockIdx.y * TM;
    const int n0 = blockIdx.x * TN;
    A  += b * aBatch;
    Bm += b * bBatch;

    const int tid  = threadIdx.x;
    const int wave = tid >> 5;
    const int lane = tid & 31;
    const int lrow = lane & 15;

    auto stage = [&](int k0, int buf) {
        // A tile: TM x TK, chunks of 8 along K
        for (int c = tid; c < ACH; c += GEMM_THREADS) {
            int i   = c >> 2;           // tile row 0..127
            int kk0 = (c & 3) << 3;     // 0,8,16,24
            int gm  = m0 + i;
            int gk0 = k0 + kk0;
            __bf16 tmp[8] __attribute__((aligned(16)));
            int base = gm * am + gk0 * ak;
            if (ak == 1 && gm < M && gk0 + 8 <= K && ((base & 7) == 0)) {
                *(uint4*)tmp = *(const uint4*)(A + base);
            } else {
                #pragma unroll
                for (int u = 0; u < 8; ++u) {
                    int gk = gk0 + u;
                    tmp[u] = (gm < M && gk < K) ? A[gm * am + gk * ak]
                                                : (__bf16)0.0f;
                }
            }
            *(uint4*)frag_slot(AsF[buf], i >> 4, i & 15, kk0) = *(uint4*)tmp;
        }
        // B tile: TK x TN, chunks of 8 along K (fixed column)
        for (int c = tid; c < BCH; c += GEMM_THREADS) {
            int j   = c >> 2;           // tile col 0..63
            int kk0 = (c & 3) << 3;
            int gn  = n0 + j;
            int gk0 = k0 + kk0;
            __bf16 tmp[8] __attribute__((aligned(16)));
            int base = gk0 * bk + gn * bn;
            if (bk == 1 && gn < Nn && gk0 + 8 <= K && ((base & 7) == 0)) {
                *(uint4*)tmp = *(const uint4*)(Bm + base);
            } else {
                #pragma unroll
                for (int u = 0; u < 8; ++u) {
                    int gk = gk0 + u;
                    tmp[u] = (gk < K && gn < Nn) ? Bm[gk * bk + gn * bn]
                                                 : (__bf16)0.0f;
                }
            }
            *(uint4*)frag_slot(BsF[buf], j >> 4, j & 15, kk0) = *(uint4*)tmp;
        }
    };

    v8f acc[2][4] = {};

    int buf = 0;
    stage(0, 0);
    __syncthreads();
    for (int k0 = 0; k0 < K; k0 += TK) {
        if (k0 + TK < K) stage(k0 + TK, buf ^ 1);   // overlap next-tile loads

        v16bf af0 = AsF[buf][wave * 32 + lane];        // rows  wave*16 ..
        v16bf af1 = AsF[buf][(4 + wave) * 32 + lane];  // rows  64+wave*16 ..
        #pragma unroll
        for (int t = 0; t < 4; ++t) {
            v16bf bfv = BsF[buf][t * 32 + lane];
            acc[0][t] = __builtin_amdgcn_wmma_f32_16x16x32_bf16(
                false, af0, false, bfv, (short)0, acc[0][t], false, false);
            acc[1][t] = __builtin_amdgcn_wmma_f32_16x16x32_bf16(
                false, af1, false, bfv, (short)0, acc[1][t], false, false);
        }
        __syncthreads();
        buf ^= 1;
    }

    // store: C/D layout -> VGPR r: lanes 0..15 M=r, lanes 16..31 M=r+8
    const int cbase = b * cBatch;
    #pragma unroll
    for (int h = 0; h < 2; ++h) {
        #pragma unroll
        for (int t = 0; t < 4; ++t) {
            int gn = n0 + t * 16 + lrow;
            #pragma unroll
            for (int r = 0; r < 8; ++r) {
                int gm = m0 + h * 64 + wave * 16 + r + (lane >> 4) * 8;
                if (gm < M && gn < Nn) {
                    float val = acc[h][t][r];
                    if (C)  C [cbase + gm * ldc + gn] = val;
                    if (Cb) Cb[cbase + gm * ldc + gn] = (__bf16)val;
                }
            }
        }
    }
}

// ---------------------------------------------------------------------------
// Implicit-GEMM 3x3 SAME conv (NCHW) with fused BN + ReLU epilogue.
//   M = Cout, K = Cin*9, N = Hh*Ww (grid.z = batch)
// Weights: OIHW flat == row-major Cout x (Cin*9)  (contiguous K -> fast path)
// ---------------------------------------------------------------------------
__global__ __launch_bounds__(GEMM_THREADS)
void k_conv3x3_wmma(const __bf16* __restrict__ X, const __bf16* __restrict__ Wgt,
                    const float* __restrict__ bn_s, const float* __restrict__ bn_b,
                    float* __restrict__ C, __bf16* __restrict__ Cb,
                    int Cout, int Cin, int Hh, int Ww)
{
    __shared__ v16bf AsF[2][8 * 32];
    __shared__ v16bf BsF[2][4 * 32];

    const int Nl = Hh * Ww;
    const int K  = Cin * 9;
    const int b  = blockIdx.z;
    const int m0 = blockIdx.y * TM;
    const int n0 = blockIdx.x * TN;
    X += b * Cin * Nl;

    const int tid  = threadIdx.x;
    const int wave = tid >> 5;
    const int lane = tid & 31;
    const int lrow = lane & 15;

    auto stage = [&](int k0, int buf) {
        for (int c = tid; c < ACH; c += GEMM_THREADS) {
            int i   = c >> 2;
            int kk0 = (c & 3) << 3;
            int gm  = m0 + i;
            int gk0 = k0 + kk0;
            __bf16 tmp[8] __attribute__((aligned(16)));
            if (gm < Cout && gk0 + 8 <= K) {
                *(uint4*)tmp = *(const uint4*)(Wgt + gm * K + gk0);
            } else {
                #pragma unroll
                for (int u = 0; u < 8; ++u) {
                    int gk = gk0 + u;
                    tmp[u] = (gm < Cout && gk < K) ? Wgt[gm * K + gk]
                                                   : (__bf16)0.0f;
                }
            }
            *(uint4*)frag_slot(AsF[buf], i >> 4, i & 15, kk0) = *(uint4*)tmp;
        }
        for (int c = tid; c < BCH; c += GEMM_THREADS) {
            int j   = c >> 2;
            int kk0 = (c & 3) << 3;
            int gn  = n0 + j;
            int gk0 = k0 + kk0;
            int yb = gn / Ww;
            int xb = gn - yb * Ww;
            __bf16 tmp[8] __attribute__((aligned(16)));
            #pragma unroll
            for (int u = 0; u < 8; ++u) {
                int gk = gk0 + u;
                __bf16 val = (__bf16)0.0f;
                if (gk < K && gn < Nl) {
                    int cin = gk / 9;
                    int t9  = gk - cin * 9;
                    int y   = yb + (t9 / 3) - 1;
                    int x   = xb + (t9 % 3) - 1;
                    if (y >= 0 && y < Hh && x >= 0 && x < Ww)
                        val = X[cin * Nl + y * Ww + x];
                }
                tmp[u] = val;
            }
            *(uint4*)frag_slot(BsF[buf], j >> 4, j & 15, kk0) = *(uint4*)tmp;
        }
    };

    v8f acc[2][4] = {};

    int buf = 0;
    stage(0, 0);
    __syncthreads();
    for (int k0 = 0; k0 < K; k0 += TK) {
        if (k0 + TK < K) stage(k0 + TK, buf ^ 1);

        v16bf af0 = AsF[buf][wave * 32 + lane];
        v16bf af1 = AsF[buf][(4 + wave) * 32 + lane];
        #pragma unroll
        for (int t = 0; t < 4; ++t) {
            v16bf bfv = BsF[buf][t * 32 + lane];
            acc[0][t] = __builtin_amdgcn_wmma_f32_16x16x32_bf16(
                false, af0, false, bfv, (short)0, acc[0][t], false, false);
            acc[1][t] = __builtin_amdgcn_wmma_f32_16x16x32_bf16(
                false, af1, false, bfv, (short)0, acc[1][t], false, false);
        }
        __syncthreads();
        buf ^= 1;
    }

    // epilogue: relu(acc * s/sqrt(1+eps) + b)
    const float rs = rsqrtf(1.0f + 1e-5f);
    const int cbase = b * Cout * Nl;
    #pragma unroll
    for (int h = 0; h < 2; ++h) {
        #pragma unroll
        for (int t = 0; t < 4; ++t) {
            int gn = n0 + t * 16 + lrow;
            #pragma unroll
            for (int r = 0; r < 8; ++r) {
                int gm = m0 + h * 64 + wave * 16 + r + (lane >> 4) * 8;
                if (gm < Cout && gn < Nl) {
                    float val = fmaxf(acc[h][t][r] * bn_s[gm] * rs + bn_b[gm], 0.0f);
                    if (C)  C [cbase + gm * Nl + gn] = val;
                    if (Cb) Cb[cbase + gm * Nl + gn] = (__bf16)val;
                }
            }
        }
    }
}

// ---------------------------------------------------------------------------
// lr[b,j,n] = sigmoid( sum_c q[b,c,n] * k[b,c,neighbor_j(n)] ), zero-pad
// ---------------------------------------------------------------------------
__global__ __launch_bounds__(256)
void k_lr(const float* __restrict__ q, const float* __restrict__ k,
          float* __restrict__ lr)
{
    int i = blockIdx.x * blockDim.x + threadIdx.x;
    if (i >= BB * 9 * NN) return;
    int n = i % NN;
    int j = (i / NN) % 9;
    int b = i / (9 * NN);
    int y = n / WW + (j / 3) - 1;
    int x = n % WW + (j % 3) - 1;
    float s = 0.0f;
    if (y >= 0 && y < HH && x >= 0 && x < WW) {
        int m = y * WW + x;
        const float* qp = q + b * CQ * NN;
        const float* kp = k + b * CQ * NN;
        for (int c = 0; c < CQ; ++c) s += qp[c * NN + n] * kp[c * NN + m];
    }
    lr[i] = 1.0f / (1.0f + expf(-s));
}

// ---------------------------------------------------------------------------
// In-place softmax over rows of length rowlen (one block per row)
// ---------------------------------------------------------------------------
__global__ __launch_bounds__(256)
void k_row_softmax(float* __restrict__ data, int rowlen)
{
    __shared__ float red[256];
    float* p = data + (long long)blockIdx.x * rowlen;
    float mx = -3.4e38f;
    for (int i = threadIdx.x; i < rowlen; i += 256) mx = fmaxf(mx, p[i]);
    red[threadIdx.x] = mx; __syncthreads();
    for (int s = 128; s > 0; s >>= 1) {
        if (threadIdx.x < s) red[threadIdx.x] = fmaxf(red[threadIdx.x], red[threadIdx.x + s]);
        __syncthreads();
    }
    mx = red[0]; __syncthreads();
    float sum = 0.0f;
    for (int i = threadIdx.x; i < rowlen; i += 256) {
        float e = expf(p[i] - mx); p[i] = e; sum += e;
    }
    red[threadIdx.x] = sum; __syncthreads();
    for (int s = 128; s > 0; s >>= 1) {
        if (threadIdx.x < s) red[threadIdx.x] += red[threadIdx.x + s];
        __syncthreads();
    }
    float inv = 1.0f / red[0];
    for (int i = threadIdx.x; i < rowlen; i += 256) p[i] *= inv;
}

// ---------------------------------------------------------------------------
// In-place per-channel bias add on (Bc, Cc, Nl)
// ---------------------------------------------------------------------------
__global__ __launch_bounds__(256)
void k_bias_add(float* __restrict__ data, const float* __restrict__ bias,
                int Cc, int Nl, int total)
{
    int i = blockIdx.x * blockDim.x + threadIdx.x;
    if (i >= total) return;
    int c = (i / Nl) % Cc;
    data[i] += bias[c];
}

// ---------------------------------------------------------------------------
// Bilinear resize (jnp.linspace endpoint mapping), (BC,Hin,Win)->(BC,Hout,Wout)
// ---------------------------------------------------------------------------
__global__ __launch_bounds__(256)
void k_bilinear(const float* __restrict__ in, float* __restrict__ out,
                int BC, int Hin, int Win, int Hout, int Wout)
{
    int total = BC * Hout * Wout;
    int i = blockIdx.x * blockDim.x + threadIdx.x;
    if (i >= total) return;
    int xo = i % Wout;
    int yo = (i / Wout) % Hout;
    int bc = i / (Wout * Hout);
    float fy = (Hout > 1) ? yo * (float)(Hin - 1) / (float)(Hout - 1) : 0.0f;
    float fx = (Wout > 1) ? xo * (float)(Win - 1) / (float)(Wout - 1) : 0.0f;
    int y0 = (int)floorf(fy); int y1 = min(y0 + 1, Hin - 1); float wy = fy - y0;
    int x0 = (int)floorf(fx); int x1 = min(x0 + 1, Win - 1); float wx = fx - x0;
    const float* p = in + bc * Hin * Win;
    float r0 = p[y0 * Win + x0] * (1.0f - wx) + p[y0 * Win + x1] * wx;
    float r1 = p[y1 * Win + x0] * (1.0f - wx) + p[y1 * Win + x1] * wx;
    out[i] = r0 * (1.0f - wy) + r1 * wy;
}

// ---------------------------------------------------------------------------
// Channel softmax over NC=19 at each (b,n); writes bf16 pred for gate GEMM
// ---------------------------------------------------------------------------
__global__ __launch_bounds__(256)
void k_pred_softmax(const float* __restrict__ in, __bf16* __restrict__ outb)
{
    int i = blockIdx.x * blockDim.x + threadIdx.x;
    if (i >= BB * NN) return;
    int n = i % NN;
    int b = i / NN;
    const float* p = in + b * NCC * NN + n;
    float mx = -3.4e38f;
    for (int c = 0; c < NCC; ++c) mx = fmaxf(mx, p[c * NN]);
    float sum = 0.0f;
    for (int c = 0; c < NCC; ++c) sum += expf(p[c * NN] - mx);
    float inv = 1.0f / sum;
    for (int c = 0; c < NCC; ++c)
        outb[(b * NCC + c) * NN + n] = (__bf16)(expf(p[c * NN] - mx) * inv);
}

// ---------------------------------------------------------------------------
// relation *= sigmoid(gate_logits)
// ---------------------------------------------------------------------------
__global__ __launch_bounds__(256)
void k_gate(float* __restrict__ rel, const float* __restrict__ gl, int n)
{
    int i = blockIdx.x * blockDim.x + threadIdx.x;
    if (i >= n) return;
    rel[i] *= 1.0f / (1.0f + expf(-gl[i]));
}

// ---------------------------------------------------------------------------
// aug_rel[b,n,m] = sum_j relation[b,n,neighbor_j(m)] * lr[b,j,m]  (bf16 out)
// ---------------------------------------------------------------------------
__global__ __launch_bounds__(256)
void k_augrel(const float* __restrict__ rel, const float* __restrict__ lr,
              __bf16* __restrict__ outb)
{
    int i = blockIdx.x * blockDim.x + threadIdx.x;
    if (i >= BB * NN * NN) return;
    int m = i % NN;
    int bn = i / NN;
    int n = bn % NN;
    int b = bn / NN;
    int ym = m / WW, xm = m % WW;
    const float* rb = rel + (b * NN + n) * NN;
    const float* lb = lr + b * 9 * NN;
    float s = 0.0f;
    #pragma unroll
    for (int j = 0; j < 9; ++j) {
        int y = ym + (j / 3) - 1;
        int x = xm + (j % 3) - 1;
        if (y >= 0 && y < HH && x >= 0 && x < WW)
            s += rb[y * WW + x] * lb[j * NN + m];
    }
    outb[i] = (__bf16)s;
}

// ---------------------------------------------------------------------------
// out48 = gamma[0] * aug + x
// ---------------------------------------------------------------------------
__global__ __launch_bounds__(256)
void k_scale_add(const float* __restrict__ aug, const float* __restrict__ x,
                 const float* __restrict__ gamma, float* __restrict__ out, int n)
{
    int i = blockIdx.x * blockDim.x + threadIdx.x;
    if (i >= n) return;
    out[i] = gamma[0] * aug[i] + x[i];
}

// ---------------------------------------------------------------------------
// Host-side orchestration
// ---------------------------------------------------------------------------
extern "C" void kernel_launch(void* const* d_in, const int* in_sizes, int n_in,
                              void* d_out, int out_size, void* d_ws, size_t ws_size,
                              hipStream_t stream)
{
    (void)in_sizes; (void)n_in; (void)out_size; (void)ws_size;
    const float* x     = (const float*)d_in[0];
    const float* x2    = (const float*)d_in[1];
    const float* w_q   = (const float*)d_in[2];
    const float* w_k   = (const float*)d_in[3];
    const float* w_v   = (const float*)d_in[4];
    const float* w_o31 = (const float*)d_in[5];
    const float* bn3s  = (const float*)d_in[6];
    const float* bn3b  = (const float*)d_in[7];
    const float* w_o32 = (const float*)d_in[8];
    const float* b_o32 = (const float*)d_in[9];
    const float* gamma = (const float*)d_in[10];
    const float* w_f   = (const float*)d_in[11];
    const float* bnfs  = (const float*)d_in[12];
    const float* bnfb  = (const float*)d_in[13];
    float* out = (float*)d_out;

    const int OUT0 = BB * CC * N2;  // out tensor size; aux2 follows

    // workspace bump allocator (256B aligned)
    size_t off = 0;
    auto alloc = [&](size_t bytes) -> char* {
        char* p = (char*)d_ws + off;
        off += (bytes + 255) & ~(size_t)255;
        return p;
    };
    __bf16* xb    = (__bf16*)alloc((size_t)BB * CC * NN * 2);
    __bf16* x2b   = (__bf16*)alloc((size_t)BB * C2 * N2 * 4);   // f32-sized: reused as up96 (f32)
    __bf16* wqb   = (__bf16*)alloc((size_t)CQ * CC * 2);
    __bf16* wkb   = (__bf16*)alloc((size_t)CQ * CC * 2);
    __bf16* wvb   = (__bf16*)alloc((size_t)CC * CC * 2);
    __bf16* w31b  = (__bf16*)alloc((size_t)CC * C2 * 9 * 2);
    __bf16* wfb   = (__bf16*)alloc((size_t)CC * CC * 9 * 2);
    __bf16* wo2b  = (__bf16*)alloc((size_t)NCC * CC * 2);
    float * qf    = (float *)alloc((size_t)BB * CQ * NN * 4);
    float * kf    = (float *)alloc((size_t)BB * CQ * NN * 4);
    __bf16* qb    = (__bf16*)alloc((size_t)BB * CQ * NN * 2);
    __bf16* kb    = (__bf16*)alloc((size_t)BB * CQ * NN * 2);
    __bf16* vb    = (__bf16*)alloc((size_t)BB * CC * NN * 2);
    float * lr    = (float *)alloc((size_t)BB * 9 * NN * 4);
    float * rel   = (float *)alloc((size_t)BB * NN * NN * 4);
    float * gatel = (float *)alloc((size_t)BB * NN * NN * 4);   // reused as aug_relb (bf16)
    __bf16* h1b   = (__bf16*)alloc((size_t)BB * CC * N2 * 2);   // reused as up96b
    float * predt = (float *)alloc((size_t)BB * NCC * NN * 4);
    __bf16* predb = (__bf16*)alloc((size_t)BB * NCC * NN * 2);
    float * aug   = (float *)alloc((size_t)BB * CC * NN * 4);
    float * out48 = (float *)alloc((size_t)BB * CC * NN * 4);
    // aliases (buffer reuse after producer consumed)
    __bf16* augrelb = (__bf16*)gatel;
    float * up96    = (float *)x2b;
    __bf16* up96b   = h1b;

    auto cdiv = [](int a, int b) { return (a + b - 1) / b; };
    auto cvt = [&](const float* src, __bf16* dst, int n) {
        k_cvt_bf16<<<dim3((unsigned)cdiv(n, 256)), dim3(256), 0, stream>>>(src, dst, n);
    };

    // 1. precision conversion
    cvt(x,     xb,   BB * CC * NN);
    cvt(x2,    x2b,  BB * C2 * N2);
    cvt(w_q,   wqb,  CQ * CC);
    cvt(w_k,   wkb,  CQ * CC);
    cvt(w_v,   wvb,  CC * CC);
    cvt(w_o31, w31b, CC * C2 * 9);
    cvt(w_f,   wfb,  CC * CC * 9);
    cvt(w_o32, wo2b, NCC * CC);

    // 2. q/k/v projections: [M x 512] * [512 x 2304] per batch
    dim3 gQ((unsigned)cdiv(NN, TN), (unsigned)cdiv(CQ, TM), BB);
    k_gemm_bf16<<<gQ, GEMM_THREADS, 0, stream>>>(wqb, xb, qf, qb, CQ, NN, CC,
        0, CC, 1, CC * NN, NN, 1, CQ * NN, NN);
    k_gemm_bf16<<<gQ, GEMM_THREADS, 0, stream>>>(wkb, xb, kf, kb, CQ, NN, CC,
        0, CC, 1, CC * NN, NN, 1, CQ * NN, NN);
    dim3 gV((unsigned)cdiv(NN, TN), (unsigned)cdiv(CC, TM), BB);
    k_gemm_bf16<<<gV, GEMM_THREADS, 0, stream>>>(wvb, xb, nullptr, vb, CC, NN, CC,
        0, CC, 1, CC * NN, NN, 1, CC * NN, NN);

    // 3. local relation lr
    k_lr<<<(unsigned)cdiv(BB * 9 * NN, 256), 256, 0, stream>>>(qf, kf, lr);

    // 4. relation logits: q^T k  (A = q transposed via strides)
    dim3 gR((unsigned)cdiv(NN, TN), (unsigned)cdiv(NN, TM), BB);
    k_gemm_bf16<<<gR, GEMM_THREADS, 0, stream>>>(qb, kb, rel, nullptr, NN, NN, CQ,
        CQ * NN, 1, NN, CQ * NN, NN, 1, NN * NN, NN);

    // 5. softmax over last dim
    k_row_softmax<<<(unsigned)(BB * NN), 256, 0, stream>>>(rel, NN);

    // 6. conv3x3 1024->512 @96x96, BN+ReLU, bf16 out
    dim3 gC1((unsigned)cdiv(N2, TN), (unsigned)cdiv(CC, TM), BB);
    k_conv3x3_wmma<<<gC1, GEMM_THREADS, 0, stream>>>(x2b, w31b, bn3s, bn3b,
        nullptr, h1b, CC, C2, H2, W2);

    // 7. aux2 = w_o3_2 * h1 (to d_out tail), then +bias
    dim3 gA2((unsigned)cdiv(N2, TN), (unsigned)cdiv(NCC, TM), BB);
    k_gemm_bf16<<<gA2, GEMM_THREADS, 0, stream>>>(wo2b, h1b, out + OUT0, nullptr,
        NCC, N2, CC, 0, CC, 1, CC * N2, N2, 1, NCC * N2, N2);
    k_bias_add<<<(unsigned)cdiv(BB * NCC * N2, 256), 256, 0, stream>>>(
        out + OUT0, b_o32, NCC, N2, BB * NCC * N2);

    // 8. pred = softmax(bilinear(aux2, 48, 48), axis=1)
    k_bilinear<<<(unsigned)cdiv(BB * NCC * NN, 256), 256, 0, stream>>>(
        out + OUT0, predt, BB * NCC, H2, W2, HH, WW);
    k_pred_softmax<<<(unsigned)cdiv(BB * NN, 256), 256, 0, stream>>>(predt, predb);

    // 9. gate logits: pred^T pred (K=19, zero-padded in tile loads)
    k_gemm_bf16<<<gR, GEMM_THREADS, 0, stream>>>(predb, predb, gatel, nullptr, NN, NN, NCC,
        NCC * NN, 1, NN, NCC * NN, NN, 1, NN * NN, NN);

    // 10. relation *= sigmoid(gate)
    k_gate<<<(unsigned)cdiv(BB * NN * NN, 256), 256, 0, stream>>>(
        rel, gatel, BB * NN * NN);

    // 11. aug_rel 9-point stencil (bf16 out; reuses gatel storage)
    k_augrel<<<(unsigned)cdiv(BB * NN * NN, 256), 256, 0, stream>>>(rel, lr, augrelb);

    // 12. aug[c,n] = sum_m v[c,m] * aug_rel[n,m]  (B transposed via strides)
    k_gemm_bf16<<<gV, GEMM_THREADS, 0, stream>>>(vb, augrelb, aug, nullptr, CC, NN, NN,
        CC * NN, NN, 1, NN * NN, 1, NN, CC * NN, NN);

    // 13. residual, upsample 48->96, convert
    k_scale_add<<<(unsigned)cdiv(BB * CC * NN, 256), 256, 0, stream>>>(
        aug, x, gamma, out48, BB * CC * NN);
    k_bilinear<<<(unsigned)cdiv(BB * CC * N2, 256), 256, 0, stream>>>(
        out48, up96, BB * CC, HH, WW, H2, W2);
    cvt(up96, up96b, BB * CC * N2);

    // 14. final conv3x3 512->512 @96x96, BN+ReLU -> d_out
    dim3 gC2((unsigned)cdiv(N2, TN), (unsigned)cdiv(CC, TM), BB);
    k_conv3x3_wmma<<<gC2, GEMM_THREADS, 0, stream>>>(up96b, wfb, bnfs, bnfb,
        out, nullptr, CC, CC, H2, W2);
}
